// GlobalAttention3D_15101105013485
// MI455X (gfx1250) — matrix operand order repaired and checked by reading, not compiled, not run on hardware
//
#include <hip/hip_runtime.h>
#include <hip/hip_bf16.h>

typedef _Float16 f16;
typedef __attribute__((ext_vector_type(16))) _Float16 v16h;
typedef __attribute__((ext_vector_type(8)))  _Float16 v8h;
typedef __attribute__((ext_vector_type(8)))  float    v8f;

#define BB   2
#define CC   256
#define GG   8
#define CPG  32          // channels per group
#define NN   4096        // 16^3 spatial positions
#define HH   8
#define HD   32
#define EPSV 1e-5f
#define SCL  0.17677669529663687f  // 1/sqrt(hd), folded into Q at projection

// ---------------------------------------------------------------- WMMA utils
static __device__ __forceinline__ v8f wmma16(v16h a, v16h b, v8f c) {
  // D = A(16x32 f16) * B(32x16 f16) + C(16x16 f32)
  return __builtin_amdgcn_wmma_f32_16x16x32_f16(false, a, false, b,
                                                (short)0, c, false, false);
}

// A operand 16x32 f16, row-major source with leading dim lda (elements).
// ISA layout: lane row m=lane&15; VGPR0-3 hold K = 8*hf..+7,
// VGPR4-7 hold K = 16+8*hf..+7 (hf = lane>>4).
static __device__ __forceinline__ v16h load_A16x32(const f16* base, int lda, int lane) {
  const int m = lane & 15, hf = lane >> 4;
  const f16* p = base + (size_t)m * lda + (hf << 3);
  v8h lo = *(const v8h*)p;
  v8h hi = *(const v8h*)(p + 16);
  v16h r;
#pragma unroll
  for (int i = 0; i < 8; ++i) { r[i] = lo[i]; r[i + 8] = hi[i]; }
  return r;
}

// B operand 32x16 f16, row-major [K,N] source with leading dim ldb.
// ISA layout: lane column n=lane&15; elements 0-7 hold K = 8*hf..+7,
// elements 8-15 hold K = 16+8*hf..+7 (hf = lane>>4).
static __device__ __forceinline__ v16h load_B32x16(const f16* base, int ldb, int lane) {
  const int n = lane & 15, hf = lane >> 4;
  const f16* p = base + (size_t)(hf << 3) * ldb + n;
  v16h r;
#pragma unroll
  for (int i = 0; i < 8; ++i) { r[i] = p[(size_t)i * ldb]; r[i + 8] = p[(size_t)(16 + i) * ldb]; }
  return r;
}

// Online-softmax for one S tile pair (C/D layout, scores pre-scaled) and
// direct store of exp() results into the per-wave LDS P buffer (row-major
// 16x32). Row of element r is 8*hf + r; reductions are half-wave butterflies.
static __device__ __forceinline__ void softmax_store(
    const v8f& s0, const v8f& s1, float* mrow, float* lrow,
    v8f& o0, v8f& o1, f16* P, int rb, int col) {
#pragma unroll
  for (int r = 0; r < 8; ++r) {
    const float a0 = s0[r], a1 = s1[r];
    float mx = fmaxf(a0, a1);
#pragma unroll
    for (int sft = 1; sft < 16; sft <<= 1)
      mx = fmaxf(mx, __shfl_xor(mx, sft, 32));
    const float nm = fmaxf(mrow[r], mx);
    const float e0 = __expf(a0 - nm), e1 = __expf(a1 - nm);
    float sum = e0 + e1;
#pragma unroll
    for (int sft = 1; sft < 16; sft <<= 1)
      sum += __shfl_xor(sum, sft, 32);
    const float alpha = __expf(mrow[r] - nm);
    lrow[r] = lrow[r] * alpha + sum;
    mrow[r] = nm;
    o0[r] *= alpha; o1[r] *= alpha;
    P[(rb + r) * 32 + col]      = (f16)e0;
    P[(rb + r) * 32 + 16 + col] = (f16)e1;
  }
}

// -------------------------------------------------------------- GroupNorm
__global__ void ga3d_gn_stats(const float* __restrict__ x, float* __restrict__ stats) {
  const int bg = blockIdx.x;                      // 0..15, group block contiguous
  const float* p = x + (size_t)bg * (CPG * (size_t)NN);
  float s = 0.f, ss = 0.f;
  const int n4 = CPG * NN / 4;
  for (int i = threadIdx.x; i < n4; i += blockDim.x) {
    float4 v = ((const float4*)p)[i];
    s  += v.x + v.y + v.z + v.w;
    ss += v.x * v.x + v.y * v.y + v.z * v.z + v.w * v.w;
  }
  __shared__ float sh0[256], sh1[256];
  sh0[threadIdx.x] = s; sh1[threadIdx.x] = ss;
  __syncthreads();
  for (int st = 128; st > 0; st >>= 1) {
    if ((int)threadIdx.x < st) {
      sh0[threadIdx.x] += sh0[threadIdx.x + st];
      sh1[threadIdx.x] += sh1[threadIdx.x + st];
    }
    __syncthreads();
  }
  if (threadIdx.x == 0) {
    const float inv = 1.0f / (float)(CPG * NN);
    float mu  = sh0[0] * inv;
    float var = sh1[0] * inv - mu * mu;
    stats[2 * bg]     = mu;
    stats[2 * bg + 1] = rsqrtf(var + EPSV);
  }
}

__global__ void ga3d_gn_apply(const float* __restrict__ x,
                              const float* __restrict__ gamma,
                              const float* __restrict__ beta,
                              const float* __restrict__ stats,
                              f16* __restrict__ xnh) {
  const int i = blockIdx.x * blockDim.x + threadIdx.x;    // per float4
  const int base = i << 2;
  const int c  = (base >> 12) & 255;
  const int bg = base >> 17;                              // (32*4096) per group
  const float mu = stats[2 * bg], rs = stats[2 * bg + 1];
  const float sg = gamma[c] * rs;
  const float tt = beta[c] - mu * sg;
  float4 v = ((const float4*)x)[i];
  xnh[base + 0] = (f16)(v.x * sg + tt);
  xnh[base + 1] = (f16)(v.y * sg + tt);
  xnh[base + 2] = (f16)(v.z * sg + tt);
  xnh[base + 3] = (f16)(v.w * sg + tt);
}

__global__ void ga3d_cvt_w(const float* __restrict__ wq, const float* __restrict__ wk,
                           const float* __restrict__ wv, const float* __restrict__ wo,
                           f16* __restrict__ dst) {
  const int i = blockIdx.x * blockDim.x + threadIdx.x;    // per float4, 65536 total
  const int m = i >> 14;                                  // 16384 float4 per matrix
  const int j = i & 16383;
  const float* src = (m == 0) ? wq : (m == 1) ? wk : (m == 2) ? wv : wo;
  float4 v = ((const float4*)src)[j];
  f16* d = dst + (size_t)m * (CC * CC) + ((size_t)j << 2);
  d[0] = (f16)v.x; d[1] = (f16)v.y; d[2] = (f16)v.z; d[3] = (f16)v.w;
}

// -------------------------------------------------------------- QKV GEMM
// Wave computes a 16(o) x 64(n) strip for Q, K, V simultaneously:
// per k-step 3 A loads + 4 B loads feed 12 WMMAs.
__global__ void __launch_bounds__(128, 2)
ga3d_qkv(const f16* __restrict__ xnh,
         const f16* __restrict__ wqh, const f16* __restrict__ wkh,
         const f16* __restrict__ wvh,
         const float* __restrict__ bq, const float* __restrict__ bk,
         const float* __restrict__ bv,
         f16* __restrict__ Qh, f16* __restrict__ Kh, f16* __restrict__ Vh) {
  const int lane  = threadIdx.x & 31;
  const int wave  = threadIdx.x >> 5;
  const int strip = blockIdx.x * 4 + wave;           // 0..63, 64 n per strip
  const int otile = blockIdx.y;
  const int b     = blockIdx.z;
  const int n0    = strip << 6;

  const f16* Xb = xnh + (size_t)b * CC * NN;
  v8f cq[4] = {}, ck[4] = {}, cv[4] = {};
#pragma unroll
  for (int c0 = 0; c0 < CC; c0 += 32) {
    v16h aq = load_A16x32(wqh + (size_t)otile * 16 * CC + c0, CC, lane);
    v16h ak = load_A16x32(wkh + (size_t)otile * 16 * CC + c0, CC, lane);
    v16h av = load_A16x32(wvh + (size_t)otile * 16 * CC + c0, CC, lane);
#pragma unroll
    for (int j = 0; j < 4; ++j) {
      v16h bx = load_B32x16(Xb + (size_t)c0 * NN + n0 + (j << 4), NN, lane);
      cq[j] = wmma16(aq, bx, cq[j]);
      ck[j] = wmma16(ak, bx, ck[j]);
      cv[j] = wmma16(av, bx, cv[j]);
    }
  }
  const int hf = lane >> 4, col = lane & 15;
  const int o0 = (otile << 4) + (hf << 3);           // 8 consecutive output channels
  const int h  = o0 >> 5;
  const int d0 = o0 & 31;

#pragma unroll
  for (int j = 0; j < 4; ++j) {
    const int n = n0 + (j << 4) + col;
    v8h qv, vv;
#pragma unroll
    for (int r = 0; r < 8; ++r) {
      qv[r] = (f16)((cq[j][r] + bq[o0 + r]) * SCL);  // pre-scale Q by 1/sqrt(hd)
      vv[r] = (f16)(cv[j][r] + bv[o0 + r]);
    }
    // Q,V : [b,h,n,hd]  (A / B operands for attention)
    *(v8h*)(Qh + ((size_t)(b * HH + h) * NN + n) * HD + d0) = qv;
    *(v8h*)(Vh + ((size_t)(b * HH + h) * NN + n) * HD + d0) = vv;
    // K : [b,h,hd,n]   (B operand for S = Q K^T)
#pragma unroll
    for (int r = 0; r < 8; ++r)
      Kh[((size_t)(b * HH + h) * HD + d0 + r) * NN + n] = (f16)(ck[j][r] + bk[o0 + r]);
  }
}

// -------------------------------------------------------------- Flash attention
// One wave owns 32 query rows (two A tiles resident in registers); K/V chunks
// are streamed once and reused by both tiles: 8 WMMAs per 32-column chunk.
// Tiles A and B are post-processed sequentially to minimize live registers,
// sharing one per-wave LDS P buffer (same-wave LDS ops are in-order).
__global__ void __launch_bounds__(128, 2)
ga3d_attn(const f16* __restrict__ Qh, const f16* __restrict__ Kh,
          const f16* __restrict__ Vh, f16* __restrict__ Oh) {
  const int lane  = threadIdx.x & 31;
  const int wave  = threadIdx.x >> 5;
  const int band  = blockIdx.x * 4 + wave;       // 0..127, 32 query rows per band
  const int bh    = blockIdx.y;
  const int b     = bh >> 3, h = bh & 7;
  const int n0    = band << 5;
  const int hf    = lane >> 4, col = lane & 15;
  const int rb    = hf << 3;

  const v16h aqA = load_A16x32(Qh + ((size_t)bh * NN + n0)      * HD, HD, lane);
  const v16h aqB = load_A16x32(Qh + ((size_t)bh * NN + n0 + 16) * HD, HD, lane);
  const f16* Kb = Kh + (size_t)bh * HD * NN;     // [hd, n]
  const f16* Vb = Vh + (size_t)bh * NN * HD;     // [n, hd]

  __shared__ f16 Pb[4][512];                     // one 16x32 P buffer per wave
  f16* P = Pb[wave];
  const f16* Pl = P + (size_t)col * 32 + (hf << 3);   // A-operand reload address

  float mA[8], lA[8], mB[8], lB[8];
#pragma unroll
  for (int r = 0; r < 8; ++r) { mA[r] = mB[r] = -1e30f; lA[r] = lB[r] = 0.f; }
  v8f oA0 = {}, oA1 = {}, oB0 = {}, oB1 = {};

  for (int m0 = 0; m0 < NN; m0 += 32) {
    v16h kb0 = load_B32x16(Kb + m0,      NN, lane);
    v16h kb1 = load_B32x16(Kb + m0 + 16, NN, lane);
    // issue V loads early so softmax VALU work hides their latency
    v16h vb0 = load_B32x16(Vb + (size_t)m0 * HD,      HD, lane);
    v16h vb1 = load_B32x16(Vb + (size_t)m0 * HD + 16, HD, lane);
    if (m0 + 32 < NN) {   // prefetch next K/V chunk (global_prefetch_b8)
      __builtin_prefetch(Kb + m0 + 32 + (size_t)lane * NN, 0, 0);
      __builtin_prefetch(Vb + (size_t)(m0 + 32 + lane) * HD, 0, 0);
    }
    v8f sA0 = {}, sA1 = {}, sB0 = {}, sB1 = {};
    sA0 = wmma16(aqA, kb0, sA0);
    sA1 = wmma16(aqA, kb1, sA1);
    sB0 = wmma16(aqB, kb0, sB0);
    sB1 = wmma16(aqB, kb1, sB1);     // kb0/kb1 dead after this point

    // ---- tile A: softmax -> LDS -> P x V
    softmax_store(sA0, sA1, mA, lA, oA0, oA1, P, rb, col);
    {
      v8h a0 = *(const v8h*)Pl;
      v8h a1 = *(const v8h*)(Pl + 16);
      v16h ap;
#pragma unroll
      for (int i = 0; i < 8; ++i) { ap[i] = a0[i]; ap[i + 8] = a1[i]; }
      oA0 = wmma16(ap, vb0, oA0);
      oA1 = wmma16(ap, vb1, oA1);
    }
    // ---- tile B: softmax -> LDS -> P x V (reuses the same P buffer)
    softmax_store(sB0, sB1, mB, lB, oB0, oB1, P, rb, col);
    {
      v8h a0 = *(const v8h*)Pl;
      v8h a1 = *(const v8h*)(Pl + 16);
      v16h ap;
#pragma unroll
      for (int i = 0; i < 8; ++i) { ap[i] = a0[i]; ap[i + 8] = a1[i]; }
      oB0 = wmma16(ap, vb0, oB0);
      oB1 = wmma16(ap, vb1, oB1);
    }
  }

  // epilogue: normalize rows, store to Oh [b, c=h*32+d, n]; rows r are
  // consecutive n -> one b128 store per (tile, d-half)
  f16* ob = Oh + ((size_t)b * CC + h * HD) * NN;
  v8h rA0, rA1, rB0, rB1;
#pragma unroll
  for (int r = 0; r < 8; ++r) {
    const float iA = 1.0f / lA[r], iB = 1.0f / lB[r];
    rA0[r] = (f16)(oA0[r] * iA);
    rA1[r] = (f16)(oA1[r] * iA);
    rB0[r] = (f16)(oB0[r] * iB);
    rB1[r] = (f16)(oB1[r] * iB);
  }
  *(v8h*)(ob + (size_t)col        * NN + n0 + (hf << 3))      = rA0;
  *(v8h*)(ob + (size_t)(col + 16) * NN + n0 + (hf << 3))      = rA1;
  *(v8h*)(ob + (size_t)col        * NN + n0 + 16 + (hf << 3)) = rB0;
  *(v8h*)(ob + (size_t)(col + 16) * NN + n0 + 16 + (hf << 3)) = rB1;
}

// -------------------------------------------------------------- Output projection
// 16(o) x 64(n) strip per wave: 1 A + 4 B loads feed 4 WMMAs per k-step.
__global__ void __launch_bounds__(128, 2)
ga3d_proj(const f16* __restrict__ Oh, const f16* __restrict__ woh,
          const float* __restrict__ bo, const float* __restrict__ x,
          float* __restrict__ out) {
  const int lane  = threadIdx.x & 31;
  const int wave  = threadIdx.x >> 5;
  const int strip = blockIdx.x * 4 + wave;
  const int otile = blockIdx.y;
  const int b     = blockIdx.z;
  const int n0    = strip << 6;

  const f16* Ob = Oh + (size_t)b * CC * NN;
  v8f acc[4] = {};
#pragma unroll
  for (int c0 = 0; c0 < CC; c0 += 32) {
    v16h aw = load_A16x32(woh + (size_t)otile * 16 * CC + c0, CC, lane);
#pragma unroll
    for (int j = 0; j < 4; ++j) {
      v16h bx = load_B32x16(Ob + (size_t)c0 * NN + n0 + (j << 4), NN, lane);
      acc[j] = wmma16(aw, bx, acc[j]);
    }
  }
  const int hf = lane >> 4, col = lane & 15;
  const int o0 = (otile << 4) + (hf << 3);
#pragma unroll
  for (int j = 0; j < 4; ++j) {
    const int n = n0 + (j << 4) + col;
#pragma unroll
    for (int r = 0; r < 8; ++r) {
      const size_t idx = ((size_t)b * CC + o0 + r) * NN + n;
      out[idx] = acc[j][r] + bo[o0 + r] + x[idx];       // residual
    }
  }
}

// -------------------------------------------------------------- launch
extern "C" void kernel_launch(void* const* d_in, const int* in_sizes, int n_in,
                              void* d_out, int out_size, void* d_ws, size_t ws_size,
                              hipStream_t stream) {
  const float* x     = (const float*)d_in[0];
  const float* gamma = (const float*)d_in[1];
  const float* beta  = (const float*)d_in[2];
  const float* Wq    = (const float*)d_in[3];
  const float* bq    = (const float*)d_in[4];
  const float* Wk    = (const float*)d_in[5];
  const float* bk    = (const float*)d_in[6];
  const float* Wv    = (const float*)d_in[7];
  const float* bv    = (const float*)d_in[8];
  const float* Wo    = (const float*)d_in[9];
  const float* bo    = (const float*)d_in[10];
  float* out = (float*)d_out;

  // workspace carve-up (f16 region starts 256B in; stats live at offset 0)
  float* stats = (float*)d_ws;
  f16* base = (f16*)((char*)d_ws + 256);
  size_t off = 0;
  f16* xnh = base + off; off += (size_t)BB * CC * NN;       // 2 Mi elems
  f16* wqh = base + off; off += (size_t)CC * CC;
  f16* wkh = base + off; off += (size_t)CC * CC;
  f16* wvh = base + off; off += (size_t)CC * CC;
  f16* woh = base + off; off += (size_t)CC * CC;
  f16* Qh  = base + off; off += (size_t)BB * HH * NN * HD;
  f16* Kh  = base + off; off += (size_t)BB * HH * NN * HD;
  f16* Vh  = base + off; off += (size_t)BB * HH * NN * HD;
  f16* Oh  = base + off; off += (size_t)BB * CC * NN;
  const size_t need = 256 + off * sizeof(f16);              // ~21.5 MB
  if (ws_size < need) return;

  ga3d_gn_stats<<<dim3(BB * GG), dim3(256), 0, stream>>>(x, stats);
  ga3d_gn_apply<<<dim3((BB * CC * NN / 4) / 256), dim3(256), 0, stream>>>(
      x, gamma, beta, stats, xnh);
  ga3d_cvt_w<<<dim3((4 * CC * CC / 4) / 256), dim3(256), 0, stream>>>(
      Wq, Wk, Wv, Wo, wqh);
  ga3d_qkv<<<dim3(NN / 64 / 4, CC / 16, BB), dim3(128), 0, stream>>>(
      xnh, wqh, wkh, wvh, bq, bk, bv, Qh, Kh, Vh);
  ga3d_attn<<<dim3(NN / 32 / 4, BB * HH), dim3(128), 0, stream>>>(Qh, Kh, Vh, Oh);
  ga3d_proj<<<dim3(NN / 64 / 4, CC / 16, BB), dim3(128), 0, stream>>>(
      Oh, woh, bo, x, out);
}
